// GNN_72662256713798
// MI455X (gfx1250) — matrix operand order repaired and checked by reading, not compiled, not run on hardware
//
#include <hip/hip_runtime.h>

typedef float  v2f  __attribute__((ext_vector_type(2)));
typedef float  v8f  __attribute__((ext_vector_type(8)));
typedef __bf16 v8bf __attribute__((ext_vector_type(8)));
typedef __bf16 v16bf __attribute__((ext_vector_type(16)));

#define PR_B   160           // f32 pair-row stride (gemm64 W tile)
#define SAB    72            // bf16 LDS stride (144B): row-pair bank stride 36 -> full coverage
constexpr int NN = 4096;     // nodes per graph
constexpr int BB = 4;        // batch
constexpr float BN_EPS_C = 1e-3f;

__device__ __forceinline__ v8f wmma_f32(v2f a, v2f b, v8f c) {
    return __builtin_amdgcn_wmma_f32_16x16x4_f32(false, a, false, b, (short)0, c,
                                                 false, false);
}
__device__ __forceinline__ v8f wmma_bf16(v16bf a, v16bf b, v8f c) {
    return __builtin_amdgcn_wmma_f32_16x16x32_bf16(false, a, false, b, (short)0, c,
                                                   false, false);
}
__device__ __forceinline__ v16bf cat16(v8bf a, v8bf b) {
    return __builtin_shufflevector(a, b, 0, 1, 2, 3, 4, 5, 6, 7,
                                         8, 9, 10, 11, 12, 13, 14, 15);
}

// ---------------- small GEMM (f32 WMMA): C[row, ccol+0..63] = op(A[row,:] @ Bm) -------------
__global__ __launch_bounds__(256) void gnn_gemm64_kernel(
    const float* __restrict__ A, int lda, int acol,
    const float* __restrict__ Bm, int K,
    const float* __restrict__ bias, int do_relu,
    float* __restrict__ C, int ldc, int ccol)
{
    __shared__ float ldsW[96 * PR_B];         // K <= 192 -> 96 pair rows

    const int tid = threadIdx.x;
    for (int item = tid; item < (K >> 1) * 16; item += 256) {
        const int pr = item >> 4;
        const int cg = (item & 15) << 2;
        const float4 e = *(const float4*)(Bm + (2 * pr) * 64 + cg);
        const float4 o = *(const float4*)(Bm + (2 * pr + 1) * 64 + cg);
        v2f* dst = (v2f*)&ldsW[pr * PR_B + cg * 2];
        dst[0] = (v2f){e.x, o.x};
        dst[1] = (v2f){e.y, o.y};
        dst[2] = (v2f){e.z, o.z};
        dst[3] = (v2f){e.w, o.w};
    }
    __syncthreads();

    const int lane = tid & 31;
    const int wave = tid >> 5;
    const int nlo  = lane & 15;
    const int koff = (lane >> 4) << 1;
    const int row0 = blockIdx.x * 128 + wave * 16;

    v8f acc[4];
#pragma unroll
    for (int c = 0; c < 4; ++c) acc[c] = (v8f){};

    const float* ap = A + (size_t)(row0 + nlo) * lda + acol + koff;
    for (int kk = 0; kk < K; kk += 4) {
        const v2f a = *(const v2f*)(ap + kk);
        const v2f* bp = (const v2f*)&ldsW[((kk + koff) >> 1) * PR_B + (nlo << 1)];
        acc[0] = wmma_f32(a, bp[0],  acc[0]);
        acc[1] = wmma_f32(a, bp[16], acc[1]);
        acc[2] = wmma_f32(a, bp[32], acc[2]);
        acc[3] = wmma_f32(a, bp[48], acc[3]);
    }

    const int rbase = row0 + ((lane >> 4) << 3);
#pragma unroll
    for (int c = 0; c < 4; ++c) {
        const int col = c * 16 + nlo;
        const float bv = bias ? bias[col] : 0.0f;
#pragma unroll
        for (int i = 0; i < 8; ++i) {
            float v = acc[c][i] + bv;
            if (do_relu) v = v > 0.0f ? v : 0.0f;
            C[(size_t)(rbase + i) * ldc + ccol + col] = v;
        }
    }
}

// ------------- split T[B,N,64] f32 (row-major) -> Th/Tl[B,64,N] bf16 (col-major) -------------
__global__ __launch_bounds__(256) void gnn_tcvt_kernel(
    const float* __restrict__ T,
    unsigned short* __restrict__ Th,
    unsigned short* __restrict__ Tl)
{
    __shared__ float st[64 * 68];
    const int b  = blockIdx.y;
    const int k0 = blockIdx.x * 64;
    const float* Tb = T + (size_t)b * NN * 64;
    const int tid = threadIdx.x;
    const int lr  = tid >> 4;
    const int lc4 = (tid & 15) << 2;
#pragma unroll
    for (int p = 0; p < 4; ++p) {
        const int row = p * 16 + lr;
        *(float4*)&st[row * 68 + lc4] = *(const float4*)(Tb + (size_t)(k0 + row) * 64 + lc4);
    }
    __syncthreads();
    const int col = tid >> 2;
    const int kq  = (tid & 3) << 4;           // 16 k values per thread
    unsigned hw[8], lw[8];
#pragma unroll
    for (int j = 0; j < 8; ++j) {
        const float x0 = st[(kq + 2 * j) * 68 + col];
        const float x1 = st[(kq + 2 * j + 1) * 68 + col];
        const unsigned u0 = __float_as_uint(x0), u1 = __float_as_uint(x1);
        const unsigned h0 = u0 & 0xFFFF0000u,  h1 = u1 & 0xFFFF0000u;
        hw[j] = (u0 >> 16) | h1;
        const float l0 = x0 - __uint_as_float(h0);
        const float l1 = x1 - __uint_as_float(h1);
        lw[j] = (__float_as_uint(l0) >> 16) | (__float_as_uint(l1) & 0xFFFF0000u);
    }
    const size_t off = ((size_t)(b * 64 + col)) * NN + k0 + kq;   // bf16 elements
    uint4* dh = (uint4*)(Th + off);
    uint4* dl = (uint4*)(Tl + off);
    dh[0] = (uint4){hw[0], hw[1], hw[2], hw[3]};
    dh[1] = (uint4){hw[4], hw[5], hw[6], hw[7]};
    dl[0] = (uint4){lw[0], lw[1], lw[2], lw[3]};
    dl[1] = (uint4){lw[4], lw[5], lw[6], lw[7]};
}

// ---- propagation: Y = adj[b] @ T[b] via bf16-split WMMA, fused bias/relu/batchnorm ----
// adj stays f32 in HBM; hi/lo bf16 split happens in the tile loader.
__global__ __launch_bounds__(256) void gnn_prop_kernel(
    const float* __restrict__ adj,            // [B,N,N] f32
    const unsigned short* __restrict__ Th,    // [B,64,N] bf16 hi, col-major
    const unsigned short* __restrict__ Tl,    // [B,64,N] bf16 lo
    const float* __restrict__ bias,
    const float* __restrict__ gamma,
    const float* __restrict__ beta,
    const float* __restrict__ mean,
    const float* __restrict__ var,
    float* __restrict__ outc,                 // [B,N,192]
    int ccol)
{
    __shared__ unsigned short ldsAh[128 * SAB];   // adj tile hi, row-major [m][k]
    __shared__ unsigned short ldsAl[128 * SAB];   // adj tile lo
    __shared__ unsigned short ldsTh[64 * SAB];    // T tile hi, col-major [col][k]
    __shared__ unsigned short ldsTl[64 * SAB];    // T tile lo

    const int b  = blockIdx.y;
    const int m0 = blockIdx.x * 128;
    const float* Ab = adj + (size_t)b * NN * NN;
    const unsigned short* ThB = Th + (size_t)b * 64 * NN;
    const unsigned short* TlB = Tl + (size_t)b * 64 * NN;

    const int tid  = threadIdx.x;
    const int lane = tid & 31;
    const int wave = tid >> 5;
    const int nlo  = lane & 15;
    const int kg   = lane >> 4;               // 0 or 1 (half-wave K group)
    const int lr   = tid >> 4;
    const int lc4  = (tid & 15) << 2;

    v8f acc[4];
#pragma unroll
    for (int c = 0; c < 4; ++c) acc[c] = (v8f){};

    const int arow = (wave * 16 + nlo) * SAB;

    for (int k0 = 0; k0 < NN; k0 += 64) {
        // ---- stage adj tile (128x64 f32) with on-the-fly hi/lo bf16 split ----
#pragma unroll
        for (int p = 0; p < 8; ++p) {
            const int row = p * 16 + lr;
            const float4 v = *(const float4*)(Ab + (size_t)(m0 + row) * NN + k0 + lc4);
            const unsigned u0 = __float_as_uint(v.x), u1 = __float_as_uint(v.y);
            const unsigned u2 = __float_as_uint(v.z), u3 = __float_as_uint(v.w);
            const unsigned h0 = u0 & 0xFFFF0000u, h1 = u1 & 0xFFFF0000u;
            const unsigned h2 = u2 & 0xFFFF0000u, h3 = u3 & 0xFFFF0000u;
            const float l0 = v.x - __uint_as_float(h0);
            const float l1 = v.y - __uint_as_float(h1);
            const float l2 = v.z - __uint_as_float(h2);
            const float l3 = v.w - __uint_as_float(h3);
            uint2 hp, lp;
            hp.x = (u0 >> 16) | h1;
            hp.y = (u2 >> 16) | h3;
            lp.x = (__float_as_uint(l0) >> 16) | (__float_as_uint(l1) & 0xFFFF0000u);
            lp.y = (__float_as_uint(l2) >> 16) | (__float_as_uint(l3) & 0xFFFF0000u);
            *(uint2*)&ldsAh[row * SAB + lc4] = hp;
            *(uint2*)&ldsAl[row * SAB + lc4] = lp;
        }
        // ---- stage T tile (64 cols x 64 k, already bf16 col-major): pure copies ----
#pragma unroll
        for (int it = 0; it < 2; ++it) {
            const int q   = tid + it * 256;   // 0..511
            const int col = q >> 3;
            const int sg  = (q & 7) << 3;     // 8 bf16 per chunk
            const size_t goff = (size_t)col * NN + k0 + sg;
            *(uint4*)&ldsTh[col * SAB + sg] = *(const uint4*)(ThB + goff);
            *(uint4*)&ldsTl[col * SAB + sg] = *(const uint4*)(TlB + goff);
        }
        __syncthreads();

        // ---- compute: 2 K-steps of 32, 4 col tiles, 3 bf16 WMMAs each ----
#pragma unroll
        for (int ks = 0; ks < 64; ks += 32) {
            const int ao = arow + ks + 8 * kg;
            const v16bf ah = cat16(*(const v8bf*)&ldsAh[ao],
                                   *(const v8bf*)&ldsAh[ao + 16]);
            const v16bf al = cat16(*(const v8bf*)&ldsAl[ao],
                                   *(const v8bf*)&ldsAl[ao + 16]);
#pragma unroll
            for (int c = 0; c < 4; ++c) {
                const int bo = (c * 16 + nlo) * SAB + ks + 8 * kg;
                const v16bf bh = cat16(*(const v8bf*)&ldsTh[bo],
                                       *(const v8bf*)&ldsTh[bo + 16]);
                const v16bf bl = cat16(*(const v8bf*)&ldsTl[bo],
                                       *(const v8bf*)&ldsTl[bo + 16]);
                acc[c] = wmma_bf16(ah, bh, acc[c]);
                acc[c] = wmma_bf16(ah, bl, acc[c]);
                acc[c] = wmma_bf16(al, bh, acc[c]);
            }
        }
        __syncthreads();
    }

    // ---- fused bias + relu + batchnorm epilogue into concat buffer ----
    float* Ob = outc + (size_t)b * NN * 192;
    const int rbase = m0 + wave * 16 + (kg << 3);
#pragma unroll
    for (int c = 0; c < 4; ++c) {
        const int col = c * 16 + nlo;
        const float bv = bias[col];
        const float sc = gamma[col] / sqrtf(var[col] + BN_EPS_C);
        const float mn = mean[col];
        const float bt = beta[col];
#pragma unroll
        for (int i = 0; i < 8; ++i) {
            float v = acc[c][i] + bv;
            v = v > 0.0f ? v : 0.0f;
            v = (v - mn) * sc + bt;
            Ob[(size_t)(rbase + i) * 192 + ccol + col] = v;
        }
    }
}

extern "C" void kernel_launch(void* const* d_in, const int* in_sizes, int n_in,
                              void* d_out, int out_size, void* d_ws, size_t ws_size,
                              hipStream_t stream) {
    (void)in_sizes; (void)n_in; (void)out_size; (void)ws_size;

    const float* x   = (const float*)d_in[0];
    const float* adj = (const float*)d_in[1];
    const float* W[3]  = {(const float*)d_in[2],  (const float*)d_in[8],  (const float*)d_in[14]};
    const float* bi[3] = {(const float*)d_in[3],  (const float*)d_in[9],  (const float*)d_in[15]};
    const float* g[3]  = {(const float*)d_in[4],  (const float*)d_in[10], (const float*)d_in[16]};
    const float* be[3] = {(const float*)d_in[5],  (const float*)d_in[11], (const float*)d_in[17]};
    const float* mm[3] = {(const float*)d_in[6],  (const float*)d_in[12], (const float*)d_in[18]};
    const float* vv[3] = {(const float*)d_in[7],  (const float*)d_in[13], (const float*)d_in[19]};
    const float* Wl = (const float*)d_in[20];
    const float* bl = (const float*)d_in[21];

    float* T   = (float*)d_ws;                            // [B*N,64]  f32 : 4 MB
    float* cat = T + (size_t)BB * NN * 64;                // [B*N,192] f32 : 12 MB
    unsigned short* Th = (unsigned short*)(cat + (size_t)BB * NN * 192);  // 2 MB
    unsigned short* Tl = Th + (size_t)BB * 64 * NN;                       // 2 MB
    float* out = (float*)d_out;

    const dim3 blk(256);
    const dim3 ggemm(BB * NN / 128);                      // 128 blocks
    const dim3 gcvt(NN / 64, BB);                         // 64 x 4 blocks
    const dim3 gprop(NN / 128, BB);                       // 32 x 4 blocks

    for (int layer = 0; layer < 3; ++layer) {
        const float* src  = (layer == 0) ? x : cat;
        const int    slda = (layer == 0) ? 64 : 192;
        const int    soff = (layer == 0) ? 0 : (layer - 1) * 64;
        gnn_gemm64_kernel<<<ggemm, blk, 0, stream>>>(src, slda, soff, W[layer], 64,
                                                     nullptr, 0, T, 64, 0);
        gnn_tcvt_kernel<<<gcvt, blk, 0, stream>>>(T, Th, Tl);
        gnn_prop_kernel<<<gprop, blk, 0, stream>>>(adj, Th, Tl, bi[layer], g[layer],
                                                   be[layer], mm[layer], vv[layer],
                                                   cat, layer * 64);
    }
    // final: out = relu(cat @ Wl + bl)
    gnn_gemm64_kernel<<<ggemm, blk, 0, stream>>>(cat, 192, 0, Wl, 192, bl, 1, out, 64, 0);
}
